// MultiHeadAttetion_88691074662906
// MI455X (gfx1250) — compile-verified, matrix operand order checked by reading, not run on hardware
//
#include <hip/hip_runtime.h>
#include <hip/hip_bf16.h>

// ---- problem constants ----
#define B_   2
#define S_   2048
#define E_   1024
#define H_   16
#define HD_  64
#define SCALE_ 0.125f   // 1/sqrt(64)

typedef float  v8f    __attribute__((ext_vector_type(8)));
typedef float  f4v    __attribute__((ext_vector_type(4)));
typedef __bf16 bf16x8  __attribute__((ext_vector_type(8)));
typedef __bf16 bf16x16 __attribute__((ext_vector_type(16)));

static __device__ inline bf16x16 join8(bf16x8 lo, bf16x8 hi) {
  return __builtin_shufflevector(lo, hi, 0,1,2,3,4,5,6,7,8,9,10,11,12,13,14,15);
}
static __device__ inline bf16x8 ld_bf8(const __bf16* p) {
  return *reinterpret_cast<const bf16x8*>(p);     // 16B aligned b128 load
}
static __device__ inline bf16x8 cvt8(const float* p) {
  f4v a = *reinterpret_cast<const f4v*>(p);
  f4v b = *reinterpret_cast<const f4v*>(p + 4);
  bf16x8 r;
#pragma unroll
  for (int i = 0; i < 4; ++i) { r[i] = (__bf16)a[i]; r[4 + i] = (__bf16)b[i]; }
  return r;
}
static __device__ inline v8f wmma_bf16(bf16x16 a, bf16x16 b, v8f c) {
  // v_wmma_f32_16x16x32_bf16: (neg_a, A, neg_b, B, c_mod, C, reuse_a, reuse_b)
  return __builtin_amdgcn_wmma_f32_16x16x32_bf16(false, a, false, b, (short)0, c,
                                                 false, false);
}

// ------------------------------------------------------------------
// Stage 0: W[K][N] fp32 -> Wt[N][K] bf16 (transposed for B-fragments)
// ------------------------------------------------------------------
__global__ void wtrans_kernel(const float* __restrict__ W, __bf16* __restrict__ Wt,
                              int K, int N) {
  int idx = blockIdx.x * blockDim.x + threadIdx.x;
  if (idx >= K * N) return;
  int n = idx / K, k = idx - n * K;
  Wt[idx] = (__bf16)W[(size_t)k * N + n];
}

// ------------------------------------------------------------------
// Stage 1: QKV projection.  One wave = 16 rows of one (b,h).
// ------------------------------------------------------------------
static __device__ inline void do_proj(const bf16x16* af, const __bf16* __restrict__ Wt,
                                      const float* __restrict__ bias,
                                      int col, int hi, v8f* acc) {
  v8f zero = {};
#pragma unroll
  for (int t = 0; t < 4; ++t) acc[t] = zero;
#pragma unroll
  for (int ks = 0; ks < 2; ++ks) {
#pragma unroll
    for (int t = 0; t < 4; ++t) {
      const __bf16* wp = Wt + (size_t)(t * 16 + col) * 64 + ks * 32 + hi * 8;
      bf16x16 bf = join8(ld_bf8(wp), ld_bf8(wp + 16));
      acc[t] = wmma_bf16(af[ks], bf, acc[t]);
    }
  }
#pragma unroll
  for (int t = 0; t < 4; ++t) {
    float bb = bias[t * 16 + col];
#pragma unroll
    for (int r = 0; r < 8; ++r) acc[t][r] += bb;
  }
}

__global__ void qkv_kernel(const float* __restrict__ x,
                           const __bf16* __restrict__ Wqt, const __bf16* __restrict__ Wkt,
                           const __bf16* __restrict__ Wvt,
                           const float* __restrict__ bq, const float* __restrict__ bk,
                           const float* __restrict__ bv,
                           __bf16* __restrict__ Q, __bf16* __restrict__ K,
                           __bf16* __restrict__ Vt) {
  const int bh = blockIdx.x;                 // 0..31
  const int b = bh >> 4, h = bh & 15;
  const int sbase = blockIdx.y * 16;
  const int lane = threadIdx.x;
  const int col = lane & 15, hi = lane >> 4;

  // A fragments: 16x64 tile of x (this head's slice), fp32 -> bf16
  bf16x16 af[2];
#pragma unroll
  for (int ks = 0; ks < 2; ++ks) {
    const float* xp = x + (size_t)(b * S_ + sbase + col) * E_ + h * HD_ + ks * 32 + hi * 8;
    af[ks] = join8(cvt8(xp), cvt8(xp + 16));
  }

  v8f acc[4];
  // Q
  do_proj(af, Wqt, bq, col, hi, acc);
#pragma unroll
  for (int t = 0; t < 4; ++t)
#pragma unroll
    for (int r = 0; r < 8; ++r)
      Q[(size_t)(bh * S_ + sbase + r + 8 * hi) * 64 + t * 16 + col] = (__bf16)acc[t][r];
  // K
  do_proj(af, Wkt, bk, col, hi, acc);
#pragma unroll
  for (int t = 0; t < 4; ++t)
#pragma unroll
    for (int r = 0; r < 8; ++r)
      K[(size_t)(bh * S_ + sbase + r + 8 * hi) * 64 + t * 16 + col] = (__bf16)acc[t][r];
  // V, stored transposed: Vt[bh][v][s]
  do_proj(af, Wvt, bv, col, hi, acc);
#pragma unroll
  for (int t = 0; t < 4; ++t)
#pragma unroll
    for (int r = 0; r < 8; ++r)
      Vt[(size_t)bh * 64 * S_ + (size_t)(t * 16 + col) * S_ + sbase + r + 8 * hi] =
          (__bf16)acc[t][r];
}

// ------------------------------------------------------------------
// Stage 2: flash-attention. One wave = 32 query rows (2 M-tiles) of one
// (b,h); streams 32 keys/iter with online softmax.  K/V B-fragments are
// loaded once and feed both M-tiles (halves the dominant L2 stream).
// ------------------------------------------------------------------
__global__ void attn_kernel(const __bf16* __restrict__ Q, const __bf16* __restrict__ K,
                            const __bf16* __restrict__ Vt, const float* __restrict__ mask,
                            __bf16* __restrict__ AO) {
  __shared__ float pbuf[16][33];
  const int bh = blockIdx.x;
  const int b = bh >> 4, h = bh & 15;
  const int qbase = blockIdx.y * 32;
  const int lane = threadIdx.x;
  const int col = lane & 15, hi = lane >> 4;

  // resident Q fragments, 2 M-tiles x 2 k-steps
  bf16x16 qf[2][2];
#pragma unroll
  for (int mt = 0; mt < 2; ++mt)
#pragma unroll
    for (int ks = 0; ks < 2; ++ks) {
      const __bf16* qp =
          Q + (size_t)(bh * S_ + qbase + mt * 16 + col) * 64 + ks * 32 + hi * 8;
      qf[mt][ks] = join8(ld_bf8(qp), ld_bf8(qp + 16));
    }

  float m[2][8], l[2][8];
  v8f O[2][4];
  v8f zero = {};
#pragma unroll
  for (int mt = 0; mt < 2; ++mt) {
#pragma unroll
    for (int r = 0; r < 8; ++r) { m[mt][r] = -1.0e30f; l[mt][r] = 0.0f; }
#pragma unroll
    for (int t = 0; t < 4; ++t) O[mt][t] = zero;
  }

  const __bf16* Kbh = K + (size_t)bh * S_ * 64;
  const __bf16* Vbh = Vt + (size_t)bh * 64 * S_;

  for (int sc = 0; sc < S_ / 32; ++sc) {
    const int sbase = sc * 32;

    // prefetch next chunk (global_prefetch_b8): K rows are 128B apart ->
    // 32 lanes cover the whole 4KB chunk; Vt rows are S_*2 B apart.
    if (sc + 1 < S_ / 32) {
      __builtin_prefetch((const void*)(Kbh + (size_t)(sbase + 32 + lane) * 64), 0, 1);
      __builtin_prefetch((const void*)(Vbh + (size_t)lane * S_ + sbase + 32), 0, 1);
      __builtin_prefetch((const void*)(Vbh + (size_t)(lane + 32) * S_ + sbase + 32), 0, 1);
    }

    // S-tiles = Q @ K^T : each K fragment feeds both M-tiles
    v8f acc[2][2];
    acc[0][0] = zero; acc[0][1] = zero; acc[1][0] = zero; acc[1][1] = zero;
#pragma unroll
    for (int t = 0; t < 2; ++t)
#pragma unroll
      for (int ks = 0; ks < 2; ++ks) {
        const __bf16* kp =
            Kbh + (size_t)(sbase + t * 16 + col) * 64 + ks * 32 + hi * 8;
        bf16x16 kf = join8(ld_bf8(kp), ld_bf8(kp + 16));
#pragma unroll
        for (int mt = 0; mt < 2; ++mt)
          acc[mt][t] = wmma_bf16(qf[mt][ks], kf, acc[mt][t]);
      }

    // per-M-tile: scale+mask, online softmax, P relayout through LDS
    bf16x16 pa[2];
#pragma unroll
    for (int mt = 0; mt < 2; ++mt) {
      float xx[2][8];
#pragma unroll
      for (int t = 0; t < 2; ++t)
#pragma unroll
        for (int r = 0; r < 8; ++r)
          xx[t][r] = acc[mt][t][r] * SCALE_ *
                     mask[(size_t)(qbase + mt * 16 + r + 8 * hi) * S_ + sbase +
                          t * 16 + col];
      float mnew[8], corr[8];
#pragma unroll
      for (int r = 0; r < 8; ++r) {
        float v = fmaxf(xx[0][r], xx[1][r]);
        v = fmaxf(v, __shfl_xor(v, 1, 16));
        v = fmaxf(v, __shfl_xor(v, 2, 16));
        v = fmaxf(v, __shfl_xor(v, 4, 16));
        v = fmaxf(v, __shfl_xor(v, 8, 16));
        mnew[r] = fmaxf(m[mt][r], v);
        corr[r] = __expf(m[mt][r] - mnew[r]);
        m[mt][r] = mnew[r];
      }
#pragma unroll
      for (int r = 0; r < 8; ++r) {
        xx[0][r] = __expf(xx[0][r] - mnew[r]);
        xx[1][r] = __expf(xx[1][r] - mnew[r]);
        float s = xx[0][r] + xx[1][r];
        s += __shfl_xor(s, 1, 16);
        s += __shfl_xor(s, 2, 16);
        s += __shfl_xor(s, 4, 16);
        s += __shfl_xor(s, 8, 16);
        l[mt][r] = l[mt][r] * corr[r] + s;
      }
#pragma unroll
      for (int t = 0; t < 4; ++t)
#pragma unroll
        for (int r = 0; r < 8; ++r) O[mt][t][r] *= corr[r];
      // C/D layout -> A layout via LDS
#pragma unroll
      for (int t = 0; t < 2; ++t)
#pragma unroll
        for (int r = 0; r < 8; ++r) pbuf[r + 8 * hi][t * 16 + col] = xx[t][r];
      __syncthreads();
#pragma unroll
      for (int i = 0; i < 8; ++i) {
        pa[mt][i]     = (__bf16)pbuf[col][hi * 8 + i];
        pa[mt][8 + i] = (__bf16)pbuf[col][16 + hi * 8 + i];
      }
      __syncthreads();
    }

    // O += P @ V : each V fragment feeds both M-tiles
#pragma unroll
    for (int t = 0; t < 4; ++t) {
      const __bf16* vp = Vbh + (size_t)(t * 16 + col) * S_ + sbase + hi * 8;
      bf16x16 vf = join8(ld_bf8(vp), ld_bf8(vp + 16));
#pragma unroll
      for (int mt = 0; mt < 2; ++mt)
        O[mt][t] = wmma_bf16(pa[mt], vf, O[mt][t]);
    }
  }
  // normalize and emit (B,S,H*64) bf16
#pragma unroll
  for (int mt = 0; mt < 2; ++mt) {
#pragma unroll
    for (int r = 0; r < 8; ++r) l[mt][r] = 1.0f / l[mt][r];
#pragma unroll
    for (int t = 0; t < 4; ++t)
#pragma unroll
      for (int r = 0; r < 8; ++r)
        AO[(size_t)(b * S_ + qbase + mt * 16 + r + 8 * hi) * E_ + h * 64 + t * 16 +
           col] = (__bf16)(O[mt][t][r] * l[mt][r]);
  }
}

// ------------------------------------------------------------------
// Stage 3: output projection (4096x1024) @ (1024x1024) + bo, fp32 out.
// 32 rows x 64 cols per wave; each Wo fragment feeds both M-tiles.
// ------------------------------------------------------------------
__global__ void proj_out_kernel(const __bf16* __restrict__ AO,
                                const __bf16* __restrict__ Wot,
                                const float* __restrict__ bo, float* __restrict__ out) {
  const int mt2 = blockIdx.x;  // 0..127 : 32-row tiles
  const int nt = blockIdx.y;   // 0..15  : 64-col groups
  const int lane = threadIdx.x;
  const int col = lane & 15, hi = lane >> 4;

  v8f acc[2][4];
  v8f zero = {};
#pragma unroll
  for (int mt = 0; mt < 2; ++mt)
#pragma unroll
    for (int t = 0; t < 4; ++t) acc[mt][t] = zero;

  for (int ks = 0; ks < 32; ++ks) {
    bf16x16 af[2];
#pragma unroll
    for (int mt = 0; mt < 2; ++mt) {
      const __bf16* ap =
          AO + (size_t)(mt2 * 32 + mt * 16 + col) * E_ + ks * 32 + hi * 8;
      af[mt] = join8(ld_bf8(ap), ld_bf8(ap + 16));
    }
#pragma unroll
    for (int t = 0; t < 4; ++t) {
      const __bf16* wp = Wot + (size_t)(nt * 64 + t * 16 + col) * E_ + ks * 32 + hi * 8;
      bf16x16 wf = join8(ld_bf8(wp), ld_bf8(wp + 16));
#pragma unroll
      for (int mt = 0; mt < 2; ++mt) acc[mt][t] = wmma_bf16(af[mt], wf, acc[mt][t]);
    }
  }
#pragma unroll
  for (int mt = 0; mt < 2; ++mt)
#pragma unroll
    for (int t = 0; t < 4; ++t) {
      float bb = bo[nt * 64 + t * 16 + col];
#pragma unroll
      for (int r = 0; r < 8; ++r)
        out[(size_t)(mt2 * 32 + mt * 16 + r + 8 * hi) * E_ + nt * 64 + t * 16 + col] =
            acc[mt][t][r] + bb;
    }
}

// ------------------------------------------------------------------
extern "C" void kernel_launch(void* const* d_in, const int* in_sizes, int n_in,
                              void* d_out, int out_size, void* d_ws, size_t ws_size,
                              hipStream_t stream) {
  const float* x    = (const float*)d_in[0];
  const float* mask = (const float*)d_in[1];
  const float* Wq   = (const float*)d_in[2];
  const float* bq   = (const float*)d_in[3];
  const float* Wk   = (const float*)d_in[4];
  const float* bk   = (const float*)d_in[5];
  const float* Wv   = (const float*)d_in[6];
  const float* bv   = (const float*)d_in[7];
  const float* Wo   = (const float*)d_in[8];
  const float* bo   = (const float*)d_in[9];
  float* out = (float*)d_out;

  char* ws = (char*)d_ws;
  size_t off = 0;
  __bf16* Wqt = (__bf16*)(ws + off); off += 64 * 64 * 2;
  __bf16* Wkt = (__bf16*)(ws + off); off += 64 * 64 * 2;
  __bf16* Wvt = (__bf16*)(ws + off); off += 64 * 64 * 2;
  __bf16* Wot = (__bf16*)(ws + off); off += (size_t)E_ * E_ * 2;
  off = (off + 255) & ~(size_t)255;
  const size_t qkvBytes = (size_t)B_ * H_ * S_ * 64 * 2;
  __bf16* Qb  = (__bf16*)(ws + off); off += qkvBytes;
  __bf16* Kb  = (__bf16*)(ws + off); off += qkvBytes;
  __bf16* Vtb = (__bf16*)(ws + off); off += qkvBytes;
  __bf16* AO  = (__bf16*)(ws + off); off += (size_t)B_ * S_ * E_ * 2;

  // stage 0: weight transpose/convert
  wtrans_kernel<<<(64 * 64 + 255) / 256, 256, 0, stream>>>(Wq, Wqt, 64, 64);
  wtrans_kernel<<<(64 * 64 + 255) / 256, 256, 0, stream>>>(Wk, Wkt, 64, 64);
  wtrans_kernel<<<(64 * 64 + 255) / 256, 256, 0, stream>>>(Wv, Wvt, 64, 64);
  wtrans_kernel<<<(E_ * E_ + 255) / 256, 256, 0, stream>>>(Wo, Wot, E_, E_);

  // stage 1: QKV projection
  qkv_kernel<<<dim3(B_ * H_, S_ / 16), 32, 0, stream>>>(x, Wqt, Wkt, Wvt, bq, bk, bv,
                                                        Qb, Kb, Vtb);
  // stage 2: streaming attention (32 queries per wave)
  attn_kernel<<<dim3(B_ * H_, S_ / 32), 32, 0, stream>>>(Qb, Kb, Vtb, mask, AO);
  // stage 3: output projection (32 rows per wave)
  proj_out_kernel<<<dim3((B_ * S_) / 32, E_ / 64), 32, 0, stream>>>(AO, Wot, bo, out);
}
